// PtrDecoder_50044958933496
// MI455X (gfx1250) — matrix operand order repaired and checked by reading, not compiled, not run on hardware
//
#include <hip/hip_runtime.h>
#include <hip/hip_bf16.h>
#include <math.h>

// ---- problem constants (from reference) ----
#define Hh   200
#define Pp   16
#define Bb   32
#define Ss   600
#define ROWS (Bb * Ss)        // 19200
#define KA   640              // 600 padded to 640 (20 * 32)
#define N1   3200             // H*P
#define K2   224              // 200 padded to 224 (7 * 32)
#define LD12 448              // [m1(0..199) | pad(200..223) | m2(224..423) | pad(424..447)]
#define K3   448              // 14 * 32

typedef __attribute__((ext_vector_type(16))) _Float16 v16h;
typedef __attribute__((ext_vector_type(8)))  _Float16 v8h;
typedef __attribute__((ext_vector_type(8)))  float    v8f;

__device__ __forceinline__ float sigm(float x) { return 1.0f / (1.0f + __expf(-x)); }

// =====================================================================
// WMMA GEMM + fused maxout over groups of 16 columns.
// One wave computes a 16(M) x 128(N) tile = 8 WMMA tiles; each 16-wide N
// tile is exactly one maxout group -> cross-lane fmax butterfly.
// __launch_bounds__(128, 1): give the allocator the full VGPR budget so
// all 8 B fragments (64 VGPRs) + A fragment stay live -> one load
// clause, one s_wait_loadcnt, 8 back-to-back WMMAs, no WAR v_nops.
// =====================================================================
__global__ void __launch_bounds__(128, 1)
k_gemm_maxout16_h(const _Float16* __restrict__ A, int lda,
                  const _Float16* __restrict__ W, int ldb,
                  const float* __restrict__ bias, int kIters,
                  _Float16* __restrict__ outH, int outLd, int outColBase)
{
    const int wave   = threadIdx.x >> 5;          // 4 waves / block
    const int lane   = threadIdx.x & 31;
    const int m0     = (blockIdx.x * 4 + wave) * 16;
    const int g0     = blockIdx.y * 8;            // group index base (16 cols per group)
    const int hs     = lane >> 4;                 // half-wave select
    const int r15    = lane & 15;

    v8f zero = {0.f,0.f,0.f,0.f,0.f,0.f,0.f,0.f};
    v8f acc[8];
#pragma unroll
    for (int g = 0; g < 8; ++g) acc[g] = zero;

    const _Float16* arow = A + (size_t)(m0 + r15) * lda + (hs ? 16 : 0);
    const _Float16* wrow[8];
#pragma unroll
    for (int g = 0; g < 8; ++g)
        wrow[g] = W + (size_t)((g0 + g) * 16 + r15) * ldb + (hs ? 16 : 0);

    for (int kt = 0; kt < kIters; ++kt) {
        const int k0 = kt * 32;
        // ---- gather all fragments for this k-slice first ----
        v8h alo = *(const v8h*)(arow + k0);
        v8h ahi = *(const v8h*)(arow + k0 + 8);
        v16h bfrag[8];
#pragma unroll
        for (int g = 0; g < 8; ++g)
            bfrag[g] = *(const v16h*)(wrow[g] + k0);
        v16h afrag = __builtin_shufflevector(alo, ahi,
            0,1,2,3,4,5,6,7,8,9,10,11,12,13,14,15);
        // ---- 8 independent WMMAs, back to back ----
#pragma unroll
        for (int g = 0; g < 8; ++g)
            acc[g] = __builtin_amdgcn_wmma_f32_16x16x32_f16(
                false, afrag, false, bfrag[g], (short)0, acc[g], false, false);
    }
    // fused bias + maxout over the 16 columns of each tile
#pragma unroll
    for (int g = 0; g < 8; ++g) {
        const float bv = bias[(g0 + g) * 16 + r15];
        float mx[8];
#pragma unroll
        for (int v = 0; v < 8; ++v) {
            float x = acc[g][v] + bv;
            x = fmaxf(x, __shfl_xor(x, 1, 32));
            x = fmaxf(x, __shfl_xor(x, 2, 32));
            x = fmaxf(x, __shfl_xor(x, 4, 32));
            x = fmaxf(x, __shfl_xor(x, 8, 32));
            mx[v] = x;
        }
        if (r15 == 0) {
            const int rbase = m0 + hs * 8;
#pragma unroll
            for (int v = 0; v < 8; ++v)
                outH[(size_t)(rbase + v) * outLd + outColBase + g0 + g] = (_Float16)mx[v];
        }
    }
}

// Same pattern, single 16-col group (W3), f32 scores output.
__global__ void __launch_bounds__(128, 1)
k_scores(const _Float16* __restrict__ A, int lda,
         const _Float16* __restrict__ W, int ldb,
         const float* __restrict__ bias, int kIters,
         float* __restrict__ out)
{
    const int wave = threadIdx.x >> 5;
    const int lane = threadIdx.x & 31;
    const int m0   = (blockIdx.x * 4 + wave) * 16;
    const int hs   = lane >> 4;
    const int r15  = lane & 15;

    v8f acc = {0.f,0.f,0.f,0.f,0.f,0.f,0.f,0.f};
    const _Float16* arow = A + (size_t)(m0 + r15) * lda + (hs ? 16 : 0);
    const _Float16* wrow = W + (size_t)(r15) * ldb + (hs ? 16 : 0);
    for (int kt = 0; kt < kIters; ++kt) {
        const int k0 = kt * 32;
        v8h alo = *(const v8h*)(arow + k0);
        v8h ahi = *(const v8h*)(arow + k0 + 8);
        v16h bfrag = *(const v16h*)(wrow + k0);
        v16h afrag = __builtin_shufflevector(alo, ahi,
            0,1,2,3,4,5,6,7,8,9,10,11,12,13,14,15);
        acc = __builtin_amdgcn_wmma_f32_16x16x32_f16(
            false, afrag, false, bfrag, (short)0, acc, false, false);
    }
    const float bv = bias[r15];
    float mx[8];
#pragma unroll
    for (int v = 0; v < 8; ++v) {
        float x = acc[v] + bv;
        x = fmaxf(x, __shfl_xor(x, 1, 32));
        x = fmaxf(x, __shfl_xor(x, 2, 32));
        x = fmaxf(x, __shfl_xor(x, 4, 32));
        x = fmaxf(x, __shfl_xor(x, 8, 32));
        mx[v] = x;
    }
    if (r15 == 0) {
        const int rbase = m0 + hs * 8;
#pragma unroll
        for (int v = 0; v < 8; ++v) out[rbase + v] = mx[v];
    }
}

// ---- one-time init kernels -----------------------------------------
__global__ void k_init_A(const float* __restrict__ enc,
                         _Float16* __restrict__ Aenc, _Float16* __restrict__ A12)
{
    const int row = blockIdx.x;
    const float* src = enc + (size_t)row * 400;
    for (int c = threadIdx.x; c < KA; c += blockDim.x)
        Aenc[(size_t)row * KA + c] = (c < 400) ? (_Float16)src[c] : (_Float16)0.f;
    for (int c = threadIdx.x; c < LD12; c += blockDim.x)
        if ((c >= 200 && c < 224) || c >= 424)
            A12[(size_t)row * LD12 + c] = (_Float16)0.f;
}

__global__ void k_w1(const float* __restrict__ W1, _Float16* __restrict__ W1h) {
    int i = blockIdx.x * blockDim.x + threadIdx.x;
    if (i >= N1 * KA) return;
    int r = i / KA, c = i % KA;
    W1h[i] = (c < 600) ? (_Float16)W1[r * 600 + c] : (_Float16)0.f;
}
__global__ void k_w2(const float* __restrict__ W2, _Float16* __restrict__ W2h) {
    int i = blockIdx.x * blockDim.x + threadIdx.x;
    if (i >= N1 * K2) return;
    int r = i / K2, c = i % K2;
    W2h[i] = (c < 200) ? (_Float16)W2[r * 200 + c] : (_Float16)0.f;
}
__global__ void k_w3(const float* __restrict__ W3, _Float16* __restrict__ W3h) {
    int i = blockIdx.x * blockDim.x + threadIdx.x;
    if (i >= 16 * K3) return;
    int r = i / K3, c = i % K3;
    float v = 0.f;
    if (c < 200)                  v = W3[r * 400 + c];
    else if (c >= 224 && c < 424) v = W3[r * 400 + (c - 24)];
    W3h[i] = (_Float16)v;
}
__global__ void k_init_state(const float* __restrict__ enc,
                             float* __restrict__ h, float* __restrict__ c,
                             float* __restrict__ ptr)
{
    for (int i = threadIdx.x; i < Bb * Hh; i += blockDim.x) { h[i] = 0.f; c[i] = 0.f; }
    for (int i = threadIdx.x; i < Bb * 400; i += blockDim.x) {
        int b = i / 400, k = i % 400;
        ptr[i] = enc[((size_t)b * Ss) * 400 + k];
    }
}

// ---- per-step glue kernels -----------------------------------------
__global__ void k_r(const float* __restrict__ ptr, const float* __restrict__ h,
                    const float* __restrict__ W_lin, const float* __restrict__ b_lin,
                    float* __restrict__ r)
{
    int i = blockIdx.x * blockDim.x + threadIdx.x;
    if (i >= Bb * Hh) return;
    int b = i / Hh, j = i % Hh;
    const float* w = W_lin + (size_t)j * 600;
    float s = b_lin[j];
    for (int k = 0; k < 400; ++k) s += ptr[b * 400 + k] * w[k];
    for (int k = 0; k < 200; ++k) s += h[b * Hh + k] * w[400 + k];
    r[i] = tanhf(s);
}

__global__ void k_fill_r(const float* __restrict__ r, _Float16* __restrict__ Aenc)
{
    const int row = blockIdx.x;           // b*S + s
    const int b = row / Ss;
    if (threadIdx.x < Hh)
        Aenc[(size_t)row * KA + 400 + threadIdx.x] = (_Float16)r[b * Hh + threadIdx.x];
}

__global__ void k_lse(const float* __restrict__ scores, float* __restrict__ lse)
{
    int s = blockIdx.x * blockDim.x + threadIdx.x;
    if (s >= Ss) return;
    float mx = -INFINITY;
    for (int b = 0; b < Bb; ++b) mx = fmaxf(mx, scores[b * Ss + s]);
    float sum = 0.f;
    for (int b = 0; b < Bb; ++b) sum += __expf(scores[b * Ss + s] - mx);
    lse[s] = mx + __logf(sum);
}

__global__ void k_out(const float* __restrict__ scores, const float* __restrict__ lse,
                      float* __restrict__ out)
{
    int i = blockIdx.x * blockDim.x + threadIdx.x;
    if (i >= ROWS) return;
    out[i] = scores[i] - lse[i % Ss];
}

__global__ void k_argmax(const float* __restrict__ scores, const float* __restrict__ lse,
                         int* __restrict__ idx)
{
    int b = threadIdx.x;
    if (b >= Bb) return;
    float best = -INFINITY; int bi = 0;
    for (int s = 0; s < Ss; ++s) {
        float v = scores[b * Ss + s] - lse[s];
        if (v > best) { best = v; bi = s; }   // first-occurrence (strict >)
    }
    idx[b] = bi;
}

__global__ void k_lstm(const float* __restrict__ enc, const int* __restrict__ idx,
                       const float* __restrict__ W_ih, const float* __restrict__ W_hh,
                       const float* __restrict__ b_ih, const float* __restrict__ b_hh,
                       float* __restrict__ h, float* __restrict__ c,
                       float* __restrict__ ptr)
{
    __shared__ float np[400];
    __shared__ float hb[Hh];
    const int b = blockIdx.x;
    const int sstar = idx[b];
    const float* src = enc + ((size_t)b * Ss + sstar) * 400;
    for (int k = threadIdx.x; k < 400; k += blockDim.x) np[k] = src[k];
    for (int k = threadIdx.x; k < Hh;  k += blockDim.x) hb[k] = h[b * Hh + k];
    __syncthreads();
    if (threadIdx.x < Hh) {
        const int j = threadIdx.x;
        float gi = b_ih[j]       + b_hh[j];
        float gf = b_ih[200 + j] + b_hh[200 + j];
        float gg = b_ih[400 + j] + b_hh[400 + j];
        float go = b_ih[600 + j] + b_hh[600 + j];
        for (int k = 0; k < 400; ++k) {
            float x = np[k];
            gi += x * W_ih[(size_t)j         * 400 + k];
            gf += x * W_ih[(size_t)(200 + j) * 400 + k];
            gg += x * W_ih[(size_t)(400 + j) * 400 + k];
            go += x * W_ih[(size_t)(600 + j) * 400 + k];
        }
        for (int k = 0; k < 200; ++k) {
            float x = hb[k];
            gi += x * W_hh[(size_t)j         * 200 + k];
            gf += x * W_hh[(size_t)(200 + j) * 200 + k];
            gg += x * W_hh[(size_t)(400 + j) * 200 + k];
            go += x * W_hh[(size_t)(600 + j) * 200 + k];
        }
        float cn = sigm(gf) * c[b * Hh + j] + sigm(gi) * tanhf(gg);
        float hn = sigm(go) * tanhf(cn);
        c[b * Hh + j] = cn;
        h[b * Hh + j] = hn;
    }
    for (int k = threadIdx.x; k < 400; k += blockDim.x) ptr[b * 400 + k] = np[k];
}

// =====================================================================
extern "C" void kernel_launch(void* const* d_in, const int* in_sizes, int n_in,
                              void* d_out, int out_size, void* d_ws, size_t ws_size,
                              hipStream_t stream)
{
    const float* enc   = (const float*)d_in[0];
    const float* W_lin = (const float*)d_in[2];
    const float* b_lin = (const float*)d_in[3];
    const float* W1    = (const float*)d_in[4];
    const float* b1    = (const float*)d_in[5];
    const float* W2    = (const float*)d_in[6];
    const float* b2    = (const float*)d_in[7];
    const float* W3    = (const float*)d_in[8];
    const float* b3    = (const float*)d_in[9];
    const float* W_ih  = (const float*)d_in[10];
    const float* W_hh  = (const float*)d_in[11];
    const float* b_ih  = (const float*)d_in[12];
    const float* b_hh  = (const float*)d_in[13];
    float* out = (float*)d_out;

    const int T = out_size / (Bb * Ss);   // == answer_lengths (static 4 in reference)

    // ---- carve workspace ----
    char* w = (char*)d_ws;
    auto take = [&](size_t bytes) { char* p = w; w += (bytes + 255) & ~(size_t)255; return p; };
    _Float16* Aenc = (_Float16*)take((size_t)ROWS * KA * 2);
    _Float16* A12  = (_Float16*)take((size_t)ROWS * LD12 * 2);
    _Float16* W1h  = (_Float16*)take((size_t)N1 * KA * 2);
    _Float16* W2h  = (_Float16*)take((size_t)N1 * K2 * 2);
    _Float16* W3h  = (_Float16*)take((size_t)16 * K3 * 2);
    float* scores  = (float*)take((size_t)ROWS * 4);
    float* lse     = (float*)take((size_t)Ss * 4);
    float* rbuf    = (float*)take((size_t)Bb * Hh * 4);
    float* ptrR    = (float*)take((size_t)Bb * 400 * 4);
    float* hbuf    = (float*)take((size_t)Bb * Hh * 4);
    float* cbuf    = (float*)take((size_t)Bb * Hh * 4);
    int*   idx     = (int*)take(128);
    (void)ws_size; (void)n_in; (void)in_sizes;

    // ---- one-time conversions / init ----
    k_init_A<<<ROWS, 256, 0, stream>>>(enc, Aenc, A12);
    k_w1<<<(N1 * KA + 255) / 256, 256, 0, stream>>>(W1, W1h);
    k_w2<<<(N1 * K2 + 255) / 256, 256, 0, stream>>>(W2, W2h);
    k_w3<<<(16 * K3 + 255) / 256, 256, 0, stream>>>(W3, W3h);
    k_init_state<<<1, 256, 0, stream>>>(enc, hbuf, cbuf, ptrR);

    const dim3 gemmGrid(ROWS / 64, 25);   // 4 waves/block * 16 rows, 25 * 8 groups
    for (int t = 0; t < T; ++t) {
        k_r<<<(Bb * Hh + 255) / 256, 256, 0, stream>>>(ptrR, hbuf, W_lin, b_lin, rbuf);
        k_fill_r<<<ROWS, 256, 0, stream>>>(rbuf, Aenc);
        // m1 = maxout(ut_r @ W1^T + b1)   -> A12[:, 0:200]
        k_gemm_maxout16_h<<<gemmGrid, 128, 0, stream>>>(Aenc, KA, W1h, KA, b1, KA / 32,
                                                        A12, LD12, 0);
        // m2 = maxout(m1 @ W2^T + b2)     -> A12[:, 224:424]
        k_gemm_maxout16_h<<<gemmGrid, 128, 0, stream>>>(A12, LD12, W2h, K2, b2, K2 / 32,
                                                        A12, LD12, 224);
        // scores = maxout([m1,m2] @ W3^T + b3)
        k_scores<<<ROWS / 64, 128, 0, stream>>>(A12, LD12, W3h, K3, b3, K3 / 32, scores);
        k_lse<<<(Ss + 255) / 256, 256, 0, stream>>>(scores, lse);
        k_out<<<(ROWS + 255) / 256, 256, 0, stream>>>(scores, lse, out + (size_t)t * ROWS);
        k_argmax<<<1, 32, 0, stream>>>(scores, lse, idx);
        k_lstm<<<Bb, 256, 0, stream>>>(enc, idx, W_ih, W_hh, b_ih, b_hh, hbuf, cbuf, ptrR);
    }
}
